// VanilaTransformer_11278584119367
// MI455X (gfx1250) — matrix operand order repaired and checked by reading, not compiled, not run on hardware
//
#include <hip/hip_runtime.h>
#include <hip/hip_bf16.h>
#include <math.h>

// ---------------------------------------------------------------------------
// Transformer encoder-decoder forward for MI455X (gfx1250, wave32, WMMA).
// All heavy lifting via v_wmma_f32_16x16x32_f16 (f32 I/O, f16 compute).
// v2: vectorized f32->f16 staging (b128 loads / b64 LDS stores) + register-
//     staged LDS double buffering so next-tile HBM loads overlap WMMA compute.
// ---------------------------------------------------------------------------

typedef __attribute__((ext_vector_type(16))) _Float16 v16h;
typedef __attribute__((ext_vector_type(4)))  _Float16 v4h;
typedef __attribute__((ext_vector_type(8)))  float    v8f;

#define BQ   2
#define TT   512
#define DD   512
#define HH   8
#define LL   6
#define FF   2048
#define VV   32000
#define HDIM 4096   // H*D

// ------------------------- WMMA GEMM -------------------------
// C[M,N] = act( alpha * A[M,K] @ B[K,N] + bias[N] )
// Block tile 128x128, K-tile 32, 256 threads = 8 waves (4 row-groups x 2
// col-groups), each wave computes 32x64 via a 2x4 grid of 16x16 WMMA tiles.
// A and B staged f32->f16 into double-buffered LDS, both stored K-contiguous
// so every 16x32 fragment is two 16B LDS reads per lane.
#define BM 128
#define BN 128
#define BK 32

__global__ __launch_bounds__(256)
void gemm_f16_wmma(const float* __restrict__ A, long long sAb, long long sAh, int lda,
                   const float* __restrict__ Bg, long long sBb, long long sBh, int ldb,
                   int transB,
                   const float* __restrict__ bias,
                   float* __restrict__ C, long long sCb, long long sCh, int ldc,
                   int K, int batchH, float alpha, int act)
{
  __shared__ __align__(16) _Float16 As[2][BM * BK];
  __shared__ __align__(16) _Float16 Bs[2][BN * BK];   // Bs[n][k]

  const int tid  = threadIdx.x;
  const int lane = tid & 31;
  const int wave = tid >> 5;
  const int half = lane >> 4;   // which 16-lane half of the wave
  const int lrow = lane & 15;
  const int wm   = wave & 3;    // 4 row-groups of 32 rows
  const int wn   = wave >> 2;   // 2 col-groups of 64 cols

  // batched strided addressing: z = b*batchH + h
  const int bz = blockIdx.z;
  const int bb = bz / batchH;
  const int hh = bz - bb * batchH;
  A  += (size_t)(bb * sAb + hh * sAh);
  Bg += (size_t)(bb * sBb + hh * sBh);
  C  += (size_t)(bb * sCb + hh * sCh);

  const int m0 = blockIdx.y * BM;
  const int n0 = blockIdx.x * BN;

  v8f acc[2][4];
  #pragma unroll
  for (int i = 0; i < 2; i++)
    #pragma unroll
    for (int j = 0; j < 4; j++)
      acc[i][j] = (v8f){};

  // ---- register staging: 4 float4 per thread per matrix per K-tile ----
  // A & transB-B map: vec v = tid + i*256 -> row = v>>3, kq = v&7 (K-contig)
  // no-trans-B map:   vec v = tid + i*256 -> kk = v>>5, nq = v&31 (n-contig)
  float4 ra[4], rb[4];

  auto load_stage = [&](int kt) {
    #pragma unroll
    for (int i = 0; i < 4; i++) {
      int v = tid + i * 256;
      int row = v >> 3, kq = v & 7;
      ra[i] = *reinterpret_cast<const float4*>(&A[(size_t)(m0 + row) * lda + kt + kq * 4]);
    }
    if (transB) {
      #pragma unroll
      for (int i = 0; i < 4; i++) {
        int v = tid + i * 256;
        int row = v >> 3, kq = v & 7;
        rb[i] = *reinterpret_cast<const float4*>(&Bg[(size_t)(n0 + row) * ldb + kt + kq * 4]);
      }
    } else {
      #pragma unroll
      for (int i = 0; i < 4; i++) {
        int v = tid + i * 256;
        int kk = v >> 5, nq = v & 31;
        rb[i] = *reinterpret_cast<const float4*>(&Bg[(size_t)(kt + kk) * ldb + n0 + nq * 4]);
      }
    }
  };

  auto store_stage = [&](int buf) {
    #pragma unroll
    for (int i = 0; i < 4; i++) {
      int v = tid + i * 256;
      int row = v >> 3, kq = v & 7;
      v4h h = { (_Float16)ra[i].x, (_Float16)ra[i].y, (_Float16)ra[i].z, (_Float16)ra[i].w };
      *reinterpret_cast<v4h*>(&As[buf][row * BK + kq * 4]) = h;
    }
    if (transB) {
      #pragma unroll
      for (int i = 0; i < 4; i++) {
        int v = tid + i * 256;
        int row = v >> 3, kq = v & 7;
        v4h h = { (_Float16)rb[i].x, (_Float16)rb[i].y, (_Float16)rb[i].z, (_Float16)rb[i].w };
        *reinterpret_cast<v4h*>(&Bs[buf][row * BK + kq * 4]) = h;
      }
    } else {
      #pragma unroll
      for (int i = 0; i < 4; i++) {
        int v = tid + i * 256;
        int kk = v >> 5, nq = v & 31;
        Bs[buf][(nq * 4 + 0) * BK + kk] = (_Float16)rb[i].x;
        Bs[buf][(nq * 4 + 1) * BK + kk] = (_Float16)rb[i].y;
        Bs[buf][(nq * 4 + 2) * BK + kk] = (_Float16)rb[i].z;
        Bs[buf][(nq * 4 + 3) * BK + kk] = (_Float16)rb[i].w;
      }
    }
  };

  auto compute = [&](int buf) {
    // A 16x32 f16 layout (ISA 7.12.2): lanes 0-15 hold K=0..7 (v0..3) and
    // K=16..23 (v4..7); lanes 16-31 hold K=8..15 and K=24..31.
    v16h af[2], bf[4];
    #pragma unroll
    for (int mi = 0; mi < 2; mi++) {
      int m = wm * 32 + mi * 16 + lrow;
      uint4* d = reinterpret_cast<uint4*>(&af[mi]);
      d[0] = *reinterpret_cast<const uint4*>(&As[buf][m * BK + 8 * half]);
      d[1] = *reinterpret_cast<const uint4*>(&As[buf][m * BK + 16 + 8 * half]);
    }
    // B 32x16: per lane column n; lanes 0-15 hold K=0..7,16..23, lanes 16-31 K=8..15,24..31.
    #pragma unroll
    for (int ni = 0; ni < 4; ni++) {
      int n = wn * 64 + ni * 16 + lrow;
      uint4* d = reinterpret_cast<uint4*>(&bf[ni]);
      d[0] = *reinterpret_cast<const uint4*>(&Bs[buf][n * BK + 8 * half]);
      d[1] = *reinterpret_cast<const uint4*>(&Bs[buf][n * BK + 16 + 8 * half]);
    }
    #pragma unroll
    for (int mi = 0; mi < 2; mi++)
      #pragma unroll
      for (int ni = 0; ni < 4; ni++)
        acc[mi][ni] = __builtin_amdgcn_wmma_f32_16x16x32_f16(
            false, af[mi], false, bf[ni], (short)0, acc[mi][ni], false, false);
  };

  // ---- software-pipelined main loop (double-buffered LDS) ----
  const int nk = K / BK;
  load_stage(0);
  store_stage(0);
  __syncthreads();
  for (int t = 0; t < nk; t++) {
    const int cur = t & 1;
    if (t + 1 < nk) load_stage((t + 1) * BK);   // HBM loads overlap WMMA below
    compute(cur);
    if (t + 1 < nk) store_stage(cur ^ 1);       // fill other buffer
    __syncthreads();
  }

  // ---- epilogue: C/D layout -> VGPR r: row = 8*half + r, col = lrow ----
  #pragma unroll
  for (int mi = 0; mi < 2; mi++) {
    int rbase = m0 + wm * 32 + mi * 16 + 8 * half;
    #pragma unroll
    for (int ni = 0; ni < 4; ni++) {
      int col = n0 + wn * 64 + ni * 16 + lrow;
      float bc = bias ? bias[col] : 0.0f;
      #pragma unroll
      for (int r = 0; r < 8; r++) {
        float vv = acc[mi][ni][r] * alpha + bc;
        if (act) vv = fmaxf(vv, 0.0f);
        C[(size_t)(rbase + r) * ldc + col] = vv;
      }
    }
  }
}

// ------------------------- softmax (optionally causal) -------------------------
__global__ __launch_bounds__(256)
void softmax_rows(float* __restrict__ S, int causal)
{
  __shared__ float red[256];
  const int row   = blockIdx.x;            // over B*H*T
  const int q     = row & (TT - 1);
  const int valid = causal ? (q + 1) : TT;
  float* p = S + (size_t)row * TT;
  const int tid = threadIdx.x;

  float m = -3.4e38f;
  for (int j = tid; j < valid; j += 256) m = fmaxf(m, p[j]);
  red[tid] = m; __syncthreads();
  for (int s = 128; s > 0; s >>= 1) { if (tid < s) red[tid] = fmaxf(red[tid], red[tid + s]); __syncthreads(); }
  m = red[0]; __syncthreads();

  float acc = 0.0f;
  for (int j = tid; j < valid; j += 256) { float e = __expf(p[j] - m); p[j] = e; acc += e; }
  red[tid] = acc; __syncthreads();
  for (int s = 128; s > 0; s >>= 1) { if (tid < s) red[tid] += red[tid + s]; __syncthreads(); }
  float inv = 1.0f / red[0];

  for (int j = tid; j < valid; j += 256) p[j] *= inv;
  for (int j = valid + tid; j < TT; j += 256) p[j] = 0.0f;
}

// ------------------------- out = LN(a + x) * g + b -------------------------
__global__ __launch_bounds__(256)
void add_layernorm(const float* __restrict__ a, const float* __restrict__ x,
                   float* __restrict__ out, const float* __restrict__ g,
                   const float* __restrict__ b)
{
  __shared__ float r1[256], r2[256];
  const int row = blockIdx.x;             // over B*T
  const int tid = threadIdx.x;
  const float* pa = a + (size_t)row * DD;
  const float* px = x + (size_t)row * DD;
  float v0 = pa[tid]       + px[tid];
  float v1 = pa[tid + 256] + px[tid + 256];
  r1[tid] = v0 + v1; r2[tid] = v0 * v0 + v1 * v1;
  __syncthreads();
  for (int s = 128; s > 0; s >>= 1) {
    if (tid < s) { r1[tid] += r1[tid + s]; r2[tid] += r2[tid + s]; }
    __syncthreads();
  }
  float mean = r1[0] * (1.0f / DD);
  float var  = r2[0] * (1.0f / DD) - mean * mean;
  float inv  = rsqrtf(var + 1e-6f);
  float* po = out + (size_t)row * DD;
  po[tid]       = (v0 - mean) * inv * g[tid]       + b[tid];
  po[tid + 256] = (v1 - mean) * inv * g[tid + 256] + b[tid + 256];
}

// ------------------------- embedding * sqrt(D) + sinusoidal PE -------------------------
__global__ __launch_bounds__(256)
void embed_pe(const int* __restrict__ ids, const float* __restrict__ emb,
              float* __restrict__ out)
{
  const int tok = blockIdx.x;            // over B*T
  const int t   = tok & (TT - 1);
  const int id  = ids[tok];
  const float scale = 22.62741699796952f;          // sqrt(512)
  for (int d = threadIdx.x; d < DD; d += 256) {
    int p = d >> 1;
    float wl = __expf((float)(2 * p) * (-9.210340371976184f / (float)DD)); // -ln(1e4)/D
    float pe = (d & 1) ? __cosf((float)t * wl) : __sinf((float)t * wl);
    out[(size_t)tok * DD + d] = emb[(size_t)id * DD + d] * scale + pe;
  }
}

// ------------------------- host-side orchestration -------------------------
static inline void launch_gemm(hipStream_t st,
    const float* A, long long sAb, long long sAh, int lda,
    const float* Bg, long long sBb, long long sBh, int ldb, int transB,
    const float* bias,
    float* C, long long sCb, long long sCh, int ldc,
    int M, int N, int K, int batches, int batchH, float alpha, int act)
{
  dim3 grid(N / BN, M / BM, batches), block(256);
  gemm_f16_wmma<<<grid, block, 0, st>>>(A, sAb, sAh, lda, Bg, sBb, sBh, ldb, transB,
                                        bias, C, sCb, sCh, ldc, K, batchH, alpha, act);
}

struct Bufs { float *henc, *hdec, *q, *k, *v, *s, *o, *f, *t; };

static void mha_block(hipStream_t st, const float* qsrc, const float* kvsrc,
                      const float* w, const float* bqkv,
                      const float* ow, const float* ob,
                      const float* g, const float* be,
                      float* io, int causal, const Bufs& ws)
{
  const int M = BQ * TT;
  const size_t WSZ = (size_t)DD * HDIM;
  // Q/K/V projections: (B*T,D) @ (D,HD) + bias
  launch_gemm(st, qsrc, 0, 0, DD, w + 0 * WSZ, 0, 0, HDIM, 0, bqkv + 0 * HDIM,
              ws.q, 0, 0, HDIM, M, HDIM, DD, 1, 1, 1.0f, 0);
  launch_gemm(st, kvsrc, 0, 0, DD, w + 1 * WSZ, 0, 0, HDIM, 0, bqkv + 1 * HDIM,
              ws.k, 0, 0, HDIM, M, HDIM, DD, 1, 1, 1.0f, 0);
  launch_gemm(st, kvsrc, 0, 0, DD, w + 2 * WSZ, 0, 0, HDIM, 0, bqkv + 2 * HDIM,
              ws.v, 0, 0, HDIM, M, HDIM, DD, 1, 1, 1.0f, 0);
  // S[b,h] = Q_bh @ K_bh^T / sqrt(D)   (16-way batched, transB)
  launch_gemm(st, ws.q, (long long)TT * HDIM, DD, HDIM,
                  ws.k, (long long)TT * HDIM, DD, HDIM, 1, nullptr,
                  ws.s, (long long)HH * TT * TT, (long long)TT * TT, TT,
                  TT, TT, DD, BQ * HH, HH, 0.04419417382415922f, 0);
  softmax_rows<<<BQ * HH * TT, 256, 0, st>>>(ws.s, causal);
  // O_bh = P @ V_bh, written directly into (B,T,H*D) concat layout
  launch_gemm(st, ws.s, (long long)HH * TT * TT, (long long)TT * TT, TT,
                  ws.v, (long long)TT * HDIM, DD, HDIM, 0, nullptr,
                  ws.o, (long long)TT * HDIM, DD, HDIM,
                  TT, DD, TT, BQ * HH, HH, 1.0f, 0);
  // output projection: (B*T,HD) @ (HD,D) + ob
  launch_gemm(st, ws.o, 0, 0, HDIM, ow, 0, 0, DD, 0, ob,
              ws.t, 0, 0, DD, M, DD, HDIM, 1, 1, 1.0f, 0);
  // io = LN(proj + io)
  add_layernorm<<<M, 256, 0, st>>>(ws.t, io, io, g, be);
}

static void ffn_block(hipStream_t st, float* io,
                      const float* w1, const float* b1,
                      const float* w2, const float* b2,
                      const float* g, const float* be, const Bufs& ws)
{
  const int M = BQ * TT;
  launch_gemm(st, io,   0, 0, DD, w1, 0, 0, FF, 0, b1, ws.f, 0, 0, FF, M, FF, DD, 1, 1, 1.0f, 1);
  launch_gemm(st, ws.f, 0, 0, FF, w2, 0, 0, DD, 0, b2, ws.t, 0, 0, DD, M, DD, FF, 1, 1, 1.0f, 0);
  add_layernorm<<<M, 256, 0, st>>>(ws.t, io, io, g, be);
}

extern "C" void kernel_launch(void* const* d_in, const int* in_sizes, int n_in,
                              void* d_out, int out_size, void* d_ws, size_t ws_size,
                              hipStream_t stream)
{
  const int*   x         = (const int*)  d_in[0];
  const int*   y         = (const int*)  d_in[1];
  const float* enc_emb   = (const float*)d_in[2];
  const float* dec_emb   = (const float*)d_in[3];
  const float* enc_qkv_w = (const float*)d_in[4];
  const float* enc_qkv_b = (const float*)d_in[5];
  const float* enc_out_w = (const float*)d_in[6];
  const float* enc_out_b = (const float*)d_in[7];
  const float* enc_ln1_g = (const float*)d_in[8];
  const float* enc_ln1_b = (const float*)d_in[9];
  const float* enc_ff_w1 = (const float*)d_in[10];
  const float* enc_ff_b1 = (const float*)d_in[11];
  const float* enc_ff_w2 = (const float*)d_in[12];
  const float* enc_ff_b2 = (const float*)d_in[13];
  const float* enc_ln2_g = (const float*)d_in[14];
  const float* enc_ln2_b = (const float*)d_in[15];
  const float* dsa_qkv_w = (const float*)d_in[16];
  const float* dsa_qkv_b = (const float*)d_in[17];
  const float* dsa_out_w = (const float*)d_in[18];
  const float* dsa_out_b = (const float*)d_in[19];
  const float* dec_ln1_g = (const float*)d_in[20];
  const float* dec_ln1_b = (const float*)d_in[21];
  const float* dca_qkv_w = (const float*)d_in[22];
  const float* dca_qkv_b = (const float*)d_in[23];
  const float* dca_out_w = (const float*)d_in[24];
  const float* dca_out_b = (const float*)d_in[25];
  const float* dec_ln2_g = (const float*)d_in[26];
  const float* dec_ln2_b = (const float*)d_in[27];
  const float* dec_ff_w1 = (const float*)d_in[28];
  const float* dec_ff_b1 = (const float*)d_in[29];
  const float* dec_ff_w2 = (const float*)d_in[30];
  const float* dec_ff_b2 = (const float*)d_in[31];
  const float* dec_ln3_g = (const float*)d_in[32];
  const float* dec_ln3_b = (const float*)d_in[33];
  const float* head_w    = (const float*)d_in[34];
  const float* head_b    = (const float*)d_in[35];

  // workspace layout (floats): ~24.7M floats (~99 MB)
  float* base = (float*)d_ws;
  Bufs ws; size_t off = 0;
  ws.henc = base + off; off += (size_t)BQ * TT * DD;
  ws.hdec = base + off; off += (size_t)BQ * TT * DD;
  ws.q    = base + off; off += (size_t)BQ * TT * HDIM;
  ws.k    = base + off; off += (size_t)BQ * TT * HDIM;
  ws.v    = base + off; off += (size_t)BQ * TT * HDIM;
  ws.s    = base + off; off += (size_t)BQ * HH * TT * TT;
  ws.o    = base + off; off += (size_t)BQ * TT * HDIM;
  ws.f    = base + off; off += (size_t)BQ * TT * FF;
  ws.t    = base + off; off += (size_t)BQ * TT * DD;

  const size_t QKVW = (size_t)3 * DD * HDIM;
  const size_t QKVB = (size_t)3 * HDIM;

  // ---------------- encoder ----------------
  embed_pe<<<BQ * TT, 256, 0, stream>>>(x, enc_emb, ws.henc);
  for (int l = 0; l < LL; l++) {
    mha_block(stream, ws.henc, ws.henc,
              enc_qkv_w + l * QKVW, enc_qkv_b + l * QKVB,
              enc_out_w + (size_t)l * HDIM * DD, enc_out_b + (size_t)l * DD,
              enc_ln1_g + (size_t)l * DD, enc_ln1_b + (size_t)l * DD,
              ws.henc, 0, ws);
    ffn_block(stream, ws.henc,
              enc_ff_w1 + (size_t)l * DD * FF, enc_ff_b1 + (size_t)l * FF,
              enc_ff_w2 + (size_t)l * FF * DD, enc_ff_b2 + (size_t)l * DD,
              enc_ln2_g + (size_t)l * DD, enc_ln2_b + (size_t)l * DD, ws);
  }

  // ---------------- decoder ----------------
  embed_pe<<<BQ * TT, 256, 0, stream>>>(y, dec_emb, ws.hdec);
  for (int l = 0; l < LL; l++) {
    mha_block(stream, ws.hdec, ws.hdec,                       // causal self-attn
              dsa_qkv_w + l * QKVW, dsa_qkv_b + l * QKVB,
              dsa_out_w + (size_t)l * HDIM * DD, dsa_out_b + (size_t)l * DD,
              dec_ln1_g + (size_t)l * DD, dec_ln1_b + (size_t)l * DD,
              ws.hdec, 1, ws);
    mha_block(stream, ws.hdec, ws.henc,                       // cross-attn
              dca_qkv_w + l * QKVW, dca_qkv_b + l * QKVB,
              dca_out_w + (size_t)l * HDIM * DD, dca_out_b + (size_t)l * DD,
              dec_ln2_g + (size_t)l * DD, dec_ln2_b + (size_t)l * DD,
              ws.hdec, 0, ws);
    ffn_block(stream, ws.hdec,
              dec_ff_w1 + (size_t)l * DD * FF, dec_ff_b1 + (size_t)l * FF,
              dec_ff_w2 + (size_t)l * FF * DD, dec_ff_b2 + (size_t)l * DD,
              dec_ln3_g + (size_t)l * DD, dec_ln3_b + (size_t)l * DD, ws);
  }

  // ---------------- LM head: logits straight to d_out ----------------
  launch_gemm(stream, ws.hdec, 0, 0, DD, head_w, 0, 0, VV, 0, head_b,
              (float*)d_out, 0, 0, VV, BQ * TT, VV, DD, 1, 1, 1.0f, 0);

  (void)in_sizes; (void)n_in; (void)out_size; (void)ws_size;
}
